// BindingGNN_56100862820559
// MI455X (gfx1250) — matrix operand order repaired and checked by reading, not compiled, not run on hardware
//
#include <hip/hip_runtime.h>
#include <hip/hip_bf16.h>

#define N_NODES  20000
#define N_EDGES  320000
#define NB       16
#define NODE_D   1280
#define HID      256
#define NLAYER   4
#define NH       4
#define DHD      64
#define EH       16
#define EFULL    (N_EDGES + N_NODES)

typedef __attribute__((ext_vector_type(16))) __bf16 v16bf;
typedef __attribute__((ext_vector_type(8)))  __bf16 v8bf;
typedef __attribute__((ext_vector_type(8)))  float  v8f;

__device__ __forceinline__ float gelu_f(float x) {
    return 0.5f * x * (1.0f + erff(x * 0.70710678118654752f));
}

__device__ __forceinline__ float wred(float v) {
    #pragma unroll
    for (int o = 16; o > 0; o >>= 1) v += __shfl_xor(v, o, 32);
    return v;
}

__device__ __forceinline__ void atomicMaxF(float* a, float v) {
    if (v >= 0.0f) atomicMax((int*)a, __float_as_int(v));
    else           atomicMin((unsigned int*)a, __float_as_uint(v));
}

// ---------------- conversion kernels ----------------
__global__ void k_cvt_bf16(const float* __restrict__ X, __bf16* __restrict__ Y, int n) {
    int t = blockIdx.x * blockDim.x + threadIdx.x;
    if (t < n) Y[t] = (__bf16)X[t];
}

__global__ void k_cvt_transpose(const float* __restrict__ W, __bf16* __restrict__ Wt,
                                int K, int M) {
    int t = blockIdx.x * blockDim.x + threadIdx.x;
    if (t >= K * M) return;
    int k = t / M, m = t % M;
    Wt[(size_t)m * K + k] = (__bf16)W[t];
}

// ---------------- WMMA GEMM: C(rows x cols) = A(rows x K, bf16) * Bt^T + bias -------------
// Bt is (cols x K) row-major (B transposed). One wave computes a 16x32 tile.
__global__ void k_gemm_bf16(const __bf16* __restrict__ A, const __bf16* __restrict__ Bt,
                            const float* __restrict__ bias, float* __restrict__ C,
                            int rows, int cols, int K) {
    const int gt   = blockIdx.x * blockDim.x + threadIdx.x;
    const int wave = gt >> 5;
    const int lane = threadIdx.x & 31;
    const int ctiles = cols >> 5;
    const int tr = wave / ctiles;
    const int tc = wave % ctiles;
    if (tr * 16 >= rows) return;

    const int lm  = lane & 15;
    const int akb = (lane < 16) ? 0 : 8;    // A K-half base (ISA 16-bit A 16x32 layout)
    const int bkb = (lane < 16) ? 0 : 8;    // B K-half base (same table order as A)
    const __bf16* Ap  = A  + (size_t)(tr * 16 + lm) * K + akb;
    const __bf16* Bp0 = Bt + (size_t)(tc * 32 + lm) * K + bkb;
    const __bf16* Bp1 = Bp0 + (size_t)16 * K;

    v8f acc0 = {};
    v8f acc1 = {};
    for (int k0 = 0; k0 < K; k0 += 32) {
        __builtin_prefetch(Ap  + k0 + 256, 0, 0);
        __builtin_prefetch(Bp0 + k0 + 256, 0, 0);
        v8bf alo = *(const v8bf*)(Ap  + k0);
        v8bf ahi = *(const v8bf*)(Ap  + k0 + 16);
        v8bf b0l = *(const v8bf*)(Bp0 + k0);
        v8bf b0h = *(const v8bf*)(Bp0 + k0 + 16);
        v8bf b1l = *(const v8bf*)(Bp1 + k0);
        v8bf b1h = *(const v8bf*)(Bp1 + k0 + 16);
        v16bf a  = __builtin_shufflevector(alo, ahi, 0,1,2,3,4,5,6,7,8,9,10,11,12,13,14,15);
        v16bf b0 = __builtin_shufflevector(b0l, b0h, 0,1,2,3,4,5,6,7,8,9,10,11,12,13,14,15);
        v16bf b1 = __builtin_shufflevector(b1l, b1h, 0,1,2,3,4,5,6,7,8,9,10,11,12,13,14,15);
        acc0 = __builtin_amdgcn_wmma_f32_16x16x32_bf16(false, a, false, b0, (short)0, acc0, false, false);
        acc1 = __builtin_amdgcn_wmma_f32_16x16x32_bf16(false, a, false, b1, (short)0, acc1, false, false);
    }
    const int rbase = tr * 16 + ((lane >> 4) << 3);   // C/D layout: lanes 16-31 hold M+8
    const int c0 = tc * 32 + lm;
    const int c1 = c0 + 16;
    const float bv0 = bias ? bias[c0] : 0.0f;
    const float bv1 = bias ? bias[c1] : 0.0f;
    #pragma unroll
    for (int v = 0; v < 8; ++v) {
        C[(size_t)(rbase + v) * cols + c0] = acc0[v] + bv0;
        C[(size_t)(rbase + v) * cols + c1] = acc1[v] + bv1;
    }
}

// ---------------- input projection epilogue: LayerNorm + GELU ----------------
__global__ void k_in_ln_gelu(const float* __restrict__ Y, const float* __restrict__ g,
                             const float* __restrict__ bb, float* __restrict__ H,
                             __bf16* __restrict__ Hb) {
    int wave = (blockIdx.x * blockDim.x + threadIdx.x) >> 5;
    int lane = threadIdx.x & 31;
    if (wave >= N_NODES) return;
    const float* y = Y + (size_t)wave * HID;
    float v[8]; float s = 0.0f;
    #pragma unroll
    for (int j = 0; j < 8; ++j) { v[j] = y[lane + 32 * j]; s += v[j]; }
    float mu = wred(s) * (1.0f / HID);
    float var = 0.0f;
    #pragma unroll
    for (int j = 0; j < 8; ++j) { float d = v[j] - mu; var += d * d; }
    var = wred(var) * (1.0f / HID);
    float inv = rsqrtf(var + 1e-5f);
    #pragma unroll
    for (int j = 0; j < 8; ++j) {
        int c = lane + 32 * j;
        float t = (v[j] - mu) * inv * g[c] + bb[c];
        t = gelu_f(t);
        H [(size_t)wave * HID + c] = t;
        Hb[(size_t)wave * HID + c] = (__bf16)t;
    }
}

// ---------------- per-layer node update: gelu(agg+b) + h, then LN ----------------
__global__ void k_node_update(const float* __restrict__ Agg, const float* __restrict__ bconv,
                              const float* __restrict__ g, const float* __restrict__ bb,
                              float* __restrict__ H, __bf16* __restrict__ Hb) {
    int wave = (blockIdx.x * blockDim.x + threadIdx.x) >> 5;
    int lane = threadIdx.x & 31;
    if (wave >= N_NODES) return;
    float v[8]; float s = 0.0f;
    #pragma unroll
    for (int j = 0; j < 8; ++j) {
        int c = lane + 32 * j;
        float t = gelu_f(Agg[(size_t)wave * HID + c] + bconv[c]) + H[(size_t)wave * HID + c];
        v[j] = t; s += t;
    }
    float mu = wred(s) * (1.0f / HID);
    float var = 0.0f;
    #pragma unroll
    for (int j = 0; j < 8; ++j) { float d = v[j] - mu; var += d * d; }
    var = wred(var) * (1.0f / HID);
    float inv = rsqrtf(var + 1e-5f);
    #pragma unroll
    for (int j = 0; j < 8; ++j) {
        int c = lane + 32 * j;
        float t = (v[j] - mu) * inv * g[c] + bb[c];
        H [(size_t)wave * HID + c] = t;
        Hb[(size_t)wave * HID + c] = (__bf16)t;
    }
}

// ---------------- edge MLP + running sum for mean ----------------
__global__ void k_edge_mlp(const float* __restrict__ EA, const float* __restrict__ W1,
                           const float* __restrict__ b1, const float* __restrict__ W2,
                           const float* __restrict__ b2, float* __restrict__ EF,
                           float* __restrict__ esum) {
    __shared__ float ssum[EH];
    if (threadIdx.x < EH) ssum[threadIdx.x] = 0.0f;
    __syncthreads();
    int e = blockIdx.x * blockDim.x + threadIdx.x;
    if (e < N_EDGES) {
        float a = EA[e];
        float h1[EH], out[EH];
        #pragma unroll
        for (int j = 0; j < EH; ++j) h1[j] = gelu_f(fmaf(a, W1[j], b1[j]));
        #pragma unroll
        for (int j = 0; j < EH; ++j) out[j] = b2[j];
        #pragma unroll
        for (int k = 0; k < EH; ++k)
            #pragma unroll
            for (int j = 0; j < EH; ++j) out[j] = fmaf(h1[k], W2[k * EH + j], out[j]);
        #pragma unroll
        for (int j = 0; j < EH; ++j) {
            EF[(size_t)e * EH + j] = out[j];
            atomicAdd(&ssum[j], out[j]);
        }
    }
    __syncthreads();
    if (threadIdx.x < EH) atomicAdd(&esum[threadIdx.x], ssum[threadIdx.x]);
}

__global__ void k_fill_mean(const float* __restrict__ esum, float* __restrict__ EF) {
    int t = blockIdx.x * blockDim.x + threadIdx.x;
    if (t >= N_NODES * EH) return;
    EF[(size_t)N_EDGES * EH + t] = esum[t & (EH - 1)] * (1.0f / N_EDGES);
}

// ---------------- attention ----------------
__global__ void k_init_attn(float* __restrict__ amax, float* __restrict__ den) {
    int t = blockIdx.x * blockDim.x + threadIdx.x;
    if (t >= N_NODES * NH) return;
    ((unsigned int*)amax)[t] = 0xFF800000u;   // -inf
    den[t] = 0.0f;
}

__global__ void k_attn_logits(const float* __restrict__ XL, const float* __restrict__ XR,
                              const float* __restrict__ EF, const float* __restrict__ We,
                              const float* __restrict__ att, const int* __restrict__ eidx,
                              float* __restrict__ logits, float* __restrict__ amax) {
    int t = blockIdx.x * blockDim.x + threadIdx.x;
    if (t >= EFULL * NH) return;
    int e = t >> 2, hd = t & 3;
    int s, d;
    if (e < N_EDGES) { s = eidx[e]; d = eidx[N_EDGES + e]; }
    else             { s = d = e - N_EDGES; }
    float e16[EH];
    #pragma unroll
    for (int k = 0; k < EH; ++k) e16[k] = EF[(size_t)e * EH + k];
    const float* xls = XL + (size_t)s * HID + hd * DHD;
    const float* xrd = XR + (size_t)d * HID + hd * DHD;
    const float* at  = att + hd * DHD;
    const float* Wec = We + hd * DHD;
    float acc = 0.0f;
    for (int dd = 0; dd < DHD; ++dd) {
        float ep = 0.0f;
        #pragma unroll
        for (int k = 0; k < EH; ++k) ep = fmaf(e16[k], Wec[k * HID + dd], ep);
        float m = xls[dd] + xrd[dd] + ep;
        m = (m > 0.0f) ? m : 0.2f * m;      // leaky_relu(0.2)
        acc = fmaf(m, at[dd], acc);
    }
    logits[t] = acc;
    atomicMaxF(&amax[d * NH + hd], acc);
}

__global__ void k_attn_exp(float* __restrict__ logits, const float* __restrict__ amax,
                           float* __restrict__ den, const int* __restrict__ eidx) {
    int t = blockIdx.x * blockDim.x + threadIdx.x;
    if (t >= EFULL * NH) return;
    int e = t >> 2, hd = t & 3;
    int d = (e < N_EDGES) ? eidx[N_EDGES + e] : (e - N_EDGES);
    float ex = expf(logits[t] - amax[d * NH + hd]);
    logits[t] = ex;
    atomicAdd(&den[d * NH + hd], ex);
}

__global__ void k_attn_agg(const float* __restrict__ logits, const float* __restrict__ den,
                           const float* __restrict__ XL, const int* __restrict__ eidx,
                           float* __restrict__ Agg) {
    int t = blockIdx.x * blockDim.x + threadIdx.x;
    if (t >= EFULL * 64) return;
    int e = t >> 6;
    int g = t & 63;
    int hd = g >> 4;
    int s, d;
    if (e < N_EDGES) { s = eidx[e]; d = eidx[N_EDGES + e]; }
    else             { s = d = e - N_EDGES; }
    float alpha = logits[(size_t)e * NH + hd] / den[d * NH + hd];
    int cb = g * 4;
    const float* xs = XL + (size_t)s * HID + cb;
    float* ad = Agg + (size_t)d * HID + cb;
    #pragma unroll
    for (int q = 0; q < 4; ++q) atomicAdd(&ad[q], alpha * xs[q]);
}

// ---------------- pooling + head ----------------
__global__ void k_init_pool(float* __restrict__ gmax, float* __restrict__ gden,
                            float* __restrict__ pooled) {
    int t = blockIdx.x * blockDim.x + threadIdx.x;
    if (t < NB) { ((unsigned int*)gmax)[t] = 0xFF800000u; gden[t] = 0.0f; }
    if (t < NB * HID) pooled[t] = 0.0f;
}

__global__ void k_gate(const float* __restrict__ G1, const float* __restrict__ Wg2,
                       const float* __restrict__ bg2, const int* __restrict__ batch,
                       float* __restrict__ gate, float* __restrict__ gmax) {
    int wave = (blockIdx.x * blockDim.x + threadIdx.x) >> 5;
    int lane = threadIdx.x & 31;
    if (wave >= N_NODES) return;
    float s = 0.0f;
    #pragma unroll
    for (int j = 0; j < 4; ++j) {
        int c = lane + 32 * j;
        s = fmaf(tanhf(G1[(size_t)wave * 128 + c]), Wg2[c], s);
    }
    s = wred(s);
    if (lane == 0) {
        float v = s + bg2[0];
        gate[wave] = v;
        atomicMaxF(&gmax[batch[wave]], v);
    }
}

__global__ void k_gate_exp(float* __restrict__ gate, const float* __restrict__ gmax,
                           float* __restrict__ gden, const int* __restrict__ batch) {
    int n = blockIdx.x * blockDim.x + threadIdx.x;
    if (n >= N_NODES) return;
    int b = batch[n];
    float w = expf(gate[n] - gmax[b]);
    gate[n] = w;
    atomicAdd(&gden[b], w);
}

__global__ void k_pool(const float* __restrict__ gate, const float* __restrict__ gden,
                       const int* __restrict__ batch, const float* __restrict__ H,
                       float* __restrict__ pooled) {
    int t = blockIdx.x * blockDim.x + threadIdx.x;
    if (t >= N_NODES * 32) return;
    int n = t >> 5, lane = t & 31;
    int b = batch[n];
    float w = gate[n] / gden[b];
    #pragma unroll
    for (int j = 0; j < 8; ++j) {
        int c = lane + 32 * j;
        atomicAdd(&pooled[(size_t)b * HID + c], w * H[(size_t)n * HID + c]);
    }
}

__global__ void k_head(const float* __restrict__ P, const float* __restrict__ Wh1,
                       const float* __restrict__ bh1, const float* __restrict__ Wh2,
                       const float* __restrict__ bh2, float* __restrict__ out) {
    int wave = threadIdx.x >> 5;
    int lane = threadIdx.x & 31;
    if (wave >= NB) return;
    const float* p = P + (size_t)wave * HID;
    float tot = 0.0f;
    #pragma unroll
    for (int jj = 0; jj < 2; ++jj) {
        int j = lane + 32 * jj;
        float acc = bh1[j];
        for (int k = 0; k < HID; ++k) acc = fmaf(p[k], Wh1[k * 64 + j], acc);
        tot = fmaf(gelu_f(acc), Wh2[j], tot);
    }
    tot = wred(tot);
    if (lane == 0) out[wave] = tot + bh2[0];
}

// ---------------- host launcher ----------------
static void launch_gemm(const __bf16* A, const __bf16* Bt, const float* bias, float* C,
                        int rows, int cols, int K, hipStream_t s) {
    int waves = (rows / 16) * (cols / 32);
    int threads = 256;
    int blocks = (waves * 32 + threads - 1) / threads;
    k_gemm_bf16<<<blocks, threads, 0, s>>>(A, Bt, bias, C, rows, cols, K);
}

#define CDIV(a, b) (((a) + (b) - 1) / (b))

extern "C" void kernel_launch(void* const* d_in, const int* in_sizes, int n_in,
                              void* d_out, int out_size, void* d_ws, size_t ws_size,
                              hipStream_t stream) {
    const float* x       = (const float*)d_in[0];
    const float* ea      = (const float*)d_in[1];
    const float* W_in    = (const float*)d_in[2];
    const float* b_in    = (const float*)d_in[3];
    const float* ln_in_g = (const float*)d_in[4];
    const float* ln_in_b = (const float*)d_in[5];
    const float* W_e1    = (const float*)d_in[6];
    const float* b_e1    = (const float*)d_in[7];
    const float* W_e2    = (const float*)d_in[8];
    const float* b_e2    = (const float*)d_in[9];
    const float* Wl      = (const float*)d_in[10];
    const float* bl      = (const float*)d_in[11];
    const float* Wr      = (const float*)d_in[12];
    const float* br      = (const float*)d_in[13];
    const float* att     = (const float*)d_in[14];
    const float* We      = (const float*)d_in[15];
    const float* bconv   = (const float*)d_in[16];
    const float* ln_g    = (const float*)d_in[17];
    const float* ln_b    = (const float*)d_in[18];
    const float* Wg1     = (const float*)d_in[19];
    const float* bg1     = (const float*)d_in[20];
    const float* Wg2     = (const float*)d_in[21];
    const float* bg2     = (const float*)d_in[22];
    const float* Wh1     = (const float*)d_in[23];
    const float* bh1     = (const float*)d_in[24];
    const float* Wh2     = (const float*)d_in[25];
    const float* bh2     = (const float*)d_in[26];
    const int*   eidx    = (const int*)d_in[27];
    const int*   batch   = (const int*)d_in[28];
    float* outp = (float*)d_out;
    (void)in_sizes; (void)n_in; (void)out_size; (void)ws_size;

    char* cur = (char*)d_ws;
    auto wsAlloc = [&](size_t bytes) -> void* {
        void* p = (void*)cur;
        cur += (bytes + 255) & ~(size_t)255;
        return p;
    };
    __bf16* x_bf   = (__bf16*)wsAlloc((size_t)N_NODES * NODE_D * 2);
    __bf16* WinT   = (__bf16*)wsAlloc((size_t)HID * NODE_D * 2);
    __bf16* WlT    = (__bf16*)wsAlloc((size_t)NLAYER * HID * HID * 2);
    __bf16* WrT    = (__bf16*)wsAlloc((size_t)NLAYER * HID * HID * 2);
    __bf16* Wg1T   = (__bf16*)wsAlloc((size_t)128 * HID * 2);
    float*  h      = (float*)wsAlloc((size_t)N_NODES * HID * 4);
    __bf16* h_bf   = (__bf16*)wsAlloc((size_t)N_NODES * HID * 2);
    float*  xl     = (float*)wsAlloc((size_t)N_NODES * HID * 4);   // also GEMM temp / gate temp
    float*  xr     = (float*)wsAlloc((size_t)N_NODES * HID * 4);
    float*  agg    = (float*)wsAlloc((size_t)N_NODES * HID * 4);
    float*  efull  = (float*)wsAlloc((size_t)EFULL * EH * 4);
    float*  esum   = (float*)wsAlloc((size_t)EH * 4);
    float*  logits = (float*)wsAlloc((size_t)EFULL * NH * 4);
    float*  amax   = (float*)wsAlloc((size_t)N_NODES * NH * 4);
    float*  den    = (float*)wsAlloc((size_t)N_NODES * NH * 4);
    float*  gate   = (float*)wsAlloc((size_t)N_NODES * 4);
    float*  gmax   = (float*)wsAlloc((size_t)NB * 4);
    float*  gden   = (float*)wsAlloc((size_t)NB * 4);
    float*  pooled = (float*)wsAlloc((size_t)NB * HID * 4);

    const int TB = 256;

    // Stage 0: precision conversion (f32 -> bf16; weights transposed for B-fragment loads)
    k_cvt_bf16<<<CDIV(N_NODES * NODE_D, TB), TB, 0, stream>>>(x, x_bf, N_NODES * NODE_D);
    k_cvt_transpose<<<CDIV(NODE_D * HID, TB), TB, 0, stream>>>(W_in, WinT, NODE_D, HID);
    for (int i = 0; i < NLAYER; ++i) {
        k_cvt_transpose<<<CDIV(HID * HID, TB), TB, 0, stream>>>(
            Wl + (size_t)i * HID * HID, WlT + (size_t)i * HID * HID, HID, HID);
        k_cvt_transpose<<<CDIV(HID * HID, TB), TB, 0, stream>>>(
            Wr + (size_t)i * HID * HID, WrT + (size_t)i * HID * HID, HID, HID);
    }
    k_cvt_transpose<<<CDIV(HID * 128, TB), TB, 0, stream>>>(Wg1, Wg1T, HID, 128);

    // Stage 1: input projection (WMMA) + LN + GELU
    launch_gemm(x_bf, WinT, b_in, xl, N_NODES, HID, NODE_D, stream);
    k_in_ln_gelu<<<CDIV(N_NODES * 32, TB), TB, 0, stream>>>(xl, ln_in_g, ln_in_b, h, h_bf);

    // Stage 2: edge MLP + mean row for self-loops
    hipMemsetAsync(esum, 0, EH * 4, stream);
    k_edge_mlp<<<CDIV(N_EDGES, TB), TB, 0, stream>>>(ea, W_e1, b_e1, W_e2, b_e2, efull, esum);
    k_fill_mean<<<CDIV(N_NODES * EH, TB), TB, 0, stream>>>(esum, efull);

    // Stage 3: GATv2 layers
    for (int i = 0; i < NLAYER; ++i) {
        launch_gemm(h_bf, WlT + (size_t)i * HID * HID, bl + (size_t)i * HID, xl,
                    N_NODES, HID, HID, stream);
        launch_gemm(h_bf, WrT + (size_t)i * HID * HID, br + (size_t)i * HID, xr,
                    N_NODES, HID, HID, stream);
        k_init_attn<<<CDIV(N_NODES * NH, TB), TB, 0, stream>>>(amax, den);
        hipMemsetAsync(agg, 0, (size_t)N_NODES * HID * 4, stream);
        k_attn_logits<<<CDIV(EFULL * NH, TB), TB, 0, stream>>>(
            xl, xr, efull, We + (size_t)i * EH * HID, att + (size_t)i * NH * DHD,
            eidx, logits, amax);
        k_attn_exp<<<CDIV(EFULL * NH, TB), TB, 0, stream>>>(logits, amax, den, eidx);
        k_attn_agg<<<CDIV(EFULL * 64, TB), TB, 0, stream>>>(logits, den, xl, eidx, agg);
        k_node_update<<<CDIV(N_NODES * 32, TB), TB, 0, stream>>>(
            agg, bconv + (size_t)i * HID, ln_g + (size_t)i * HID, ln_b + (size_t)i * HID,
            h, h_bf);
    }

    // Stage 4: attention pooling (gate GEMM via WMMA) + head
    launch_gemm(h_bf, Wg1T, bg1, xl, N_NODES, 128, HID, stream);   // g1 -> xl
    k_init_pool<<<CDIV(NB * HID, TB), TB, 0, stream>>>(gmax, gden, pooled);
    k_gate<<<CDIV(N_NODES * 32, TB), TB, 0, stream>>>(xl, Wg2, bg2, batch, gate, gmax);
    k_gate_exp<<<CDIV(N_NODES, TB), TB, 0, stream>>>(gate, gmax, gden, batch);
    k_pool<<<CDIV(N_NODES * 32, TB), TB, 0, stream>>>(gate, gden, batch, h, pooled);
    k_head<<<1, NB * 32, 0, stream>>>(pooled, Wh1, bh1, Wh2, bh2, outp);
}
